// OneBranchDH_SFNN_2637109920466
// MI455X (gfx1250) — compile-verified, matrix operand order checked by reading, not run on hardware
//
#include <hip/hip_runtime.h>

// ---------------------------------------------------------------------------
// Problem constants (from the reference): B=32, T=2048, I=512, H=1024, O=1
// ---------------------------------------------------------------------------
#define BB   32
#define TT   2048
#define II   512
#define HH   1024
#define MTOT (BB * TT)      // 65536 GEMM rows
#define NCH  32             // H-chunks for the recurrence pass (32 h per wave)

typedef __attribute__((ext_vector_type(16))) _Float16 v16h;
typedef __attribute__((ext_vector_type(8)))  _Float16 v8h;
typedef __attribute__((ext_vector_type(4)))  _Float16 v4h;
typedef __attribute__((ext_vector_type(8)))  float    v8f;

// ---------------------------------------------------------------------------
// Kernel 0: fp32 -> f16 conversion (vectorized: float4 in, 4x f16 out)
// ---------------------------------------------------------------------------
__global__ void cvt_f32_to_f16_x4(const float4* __restrict__ src,
                                  _Float16* __restrict__ dst, int n4) {
  int i = blockIdx.x * blockDim.x + threadIdx.x;
  if (i < n4) {
    float4 v = src[i];
    v4h o;
    o[0] = (_Float16)v.x; o[1] = (_Float16)v.y;
    o[2] = (_Float16)v.z; o[3] = (_Float16)v.w;
    *(v4h*)(dst + (size_t)i * 4) = o;
  }
}

// ---------------------------------------------------------------------------
// Fragment loader for the 16-bit 16x32 A layout (and mirrored B layout):
// lane = row (0..15), lane bit4 selects K-half; per lane the fragment is two
// contiguous 8-halfword runs -> two global_load_b128.
//   elements 0..7  : K = kbase + 8*khalf + 0..7
//   elements 8..15 : K = kbase + 16 + 8*khalf + 0..7
// ---------------------------------------------------------------------------
__device__ __forceinline__ v16h ld_frag(const _Float16* __restrict__ p) {
  v8h lo = *(const v8h*)(p);
  v8h hi = *(const v8h*)(p + 16);
  v16h r;
#pragma unroll
  for (int i = 0; i < 8; ++i) { r[i] = lo[i]; r[i + 8] = hi[i]; }
  return r;
}

// ---------------------------------------------------------------------------
// Kernel 1: d_in[m, h] = x[m, :] . W_dense[h, :] + b_dense[h]   (f16 out)
// One wave = 32x64 output tile = 2(M) x 4(N) register-blocked WMMA tiles.
// ---------------------------------------------------------------------------
__global__ void __launch_bounds__(256)
gemm_wmma_f16(const _Float16* __restrict__ xh,    // [MTOT, II]
              const _Float16* __restrict__ wh,    // [HH, II]
              const float*    __restrict__ b_dense,
              _Float16*       __restrict__ dinh)  // [MTOT, HH]
{
  const int lane  = threadIdx.x & 31;
  const int gwave = (blockIdx.x * blockDim.x + threadIdx.x) >> 5;
  const int ntile = gwave & 15;          // 16 super-tiles of 64 cols
  const int mtile = gwave >> 4;          // 2048 super-tiles of 32 rows
  const int l15   = lane & 15;           // M (for A) / N (for B) within tile
  const int khalf = lane >> 4;           // K-half select
  const int mbase = mtile * 32;
  const int nbase = ntile * 64;

  v8f c[2][4] = {};

  const _Float16* a0 = xh + (size_t)(mbase + l15) * II;        // rows m..m+15
  const _Float16* a1 = a0 + (size_t)16 * II;                    // rows +16
  const _Float16* w0 = wh + (size_t)(nbase + l15) * II;         // W row = out col
  const int koff = 8 * khalf;

  for (int kb = 0; kb < II; kb += 32) {
    v16h a[2], b[4];
    a[0] = ld_frag(a0 + kb + koff);
    a[1] = ld_frag(a1 + kb + koff);
#pragma unroll
    for (int j = 0; j < 4; ++j)
      b[j] = ld_frag(w0 + (size_t)(j * 16) * II + kb + koff);

#pragma unroll
    for (int i = 0; i < 2; ++i)
#pragma unroll
      for (int j = 0; j < 4; ++j)
        c[i][j] = __builtin_amdgcn_wmma_f32_16x16x32_f16(
            /*neg_a=*/false, a[i], /*neg_b=*/false, b[j],
            /*c_mod=*/(short)0, c[i][j], /*reuse_a=*/false, /*reuse_b=*/false);
  }

  // C/D layout: lane&15 = N, (lane>>4)*8 + r = M within each 16x16 tile.
  const int rlo = khalf * 8;
#pragma unroll
  for (int j = 0; j < 4; ++j) {
    const int n = nbase + j * 16 + l15;
    const float bias = b_dense[n];
#pragma unroll
    for (int i = 0; i < 2; ++i) {
      size_t base = (size_t)(mbase + i * 16 + rlo) * HH + n;
#pragma unroll
      for (int r = 0; r < 8; ++r)
        dinh[base + (size_t)r * HH] = (_Float16)(c[i][j][r] + bias);
    }
  }
}

// ---------------------------------------------------------------------------
// Kernel 2: serial-in-T double EMA + partial dot with W_out.
// One wave per (batch b, 32-wide h-chunk): 1024 independent waves, no
// barriers — reduction over the 32 lanes via shfl_xor each timestep.
// ---------------------------------------------------------------------------
__global__ void __launch_bounds__(32)
recurrence(const _Float16* __restrict__ dinh,   // [MTOT, HH]
           const float* __restrict__ tau_n, const float* __restrict__ tau_m,
           const float* __restrict__ W_out,    // [1, HH]
           float* __restrict__ partial)        // [BB*NCH, TT]
{
  const int lane = threadIdx.x;          // 0..31
  const int c    = blockIdx.x & (NCH - 1);
  const int b    = blockIdx.x >> 5;      // NCH == 32
  const int h    = c * 32 + lane;

  const float beta  = 1.0f / (1.0f + expf(-tau_n[h]));
  const float alpha = 1.0f / (1.0f + expf(-tau_m[h]));
  const float w     = W_out[h];

  const _Float16* p  = dinh + (size_t)b * TT * HH + h;
  float*          po = partial + (size_t)blockIdx.x * TT;

  float dc = 0.0f, mem = 0.0f;
  for (int t = 0; t < TT; ++t) {
    if (t + 16 < TT)
      __builtin_prefetch(p + (size_t)(t + 16) * HH, 0, 0);  // global_prefetch_b8
    float din = (float)p[(size_t)t * HH];
    dc  = beta  * dc  + (1.0f - beta)  * din;
    mem = alpha * mem + (1.0f - alpha) * dc;
    float v = mem * w;
#pragma unroll
    for (int off = 16; off; off >>= 1) v += __shfl_xor(v, off, 32);
    if (lane == 0) po[t] = v;
  }
}

// ---------------------------------------------------------------------------
// Kernel 3: out[b,t] = sigmoid(b_out + sum_c partial[b,c,t])
// ---------------------------------------------------------------------------
__global__ void __launch_bounds__(256)
finalize(const float* __restrict__ partial, const float* __restrict__ b_out,
         float* __restrict__ out)
{
  int m = blockIdx.x * blockDim.x + threadIdx.x;   // m = b*TT + t
  if (m >= MTOT) return;
  int b = m / TT, t = m - b * TT;
  const float* pp = partial + (size_t)b * NCH * TT + t;
  float s = b_out[0];
#pragma unroll
  for (int cc = 0; cc < NCH; ++cc) s += pp[(size_t)cc * TT];
  out[m] = 1.0f / (1.0f + expf(-s));
}

// ---------------------------------------------------------------------------
// Host-side launcher. Workspace layout (bytes):
//   [0, 64MB)        xh    : f16 x           (MTOT*II*2  = 64 MB)
//   [64MB, 65MB)     wh    : f16 W_dense     (HH*II*2    =  1 MB)
//   [65MB, 193MB)    dinh  : f16 d_in        (MTOT*HH*2  = 128 MB)
//   [193MB, 201MB)   partial: f32            (BB*NCH*TT*4 =  8 MB)
// ---------------------------------------------------------------------------
extern "C" void kernel_launch(void* const* d_in, const int* in_sizes, int n_in,
                              void* d_out, int out_size, void* d_ws, size_t ws_size,
                              hipStream_t stream) {
  const float* x       = (const float*)d_in[0];
  const float* W_dense = (const float*)d_in[1];
  const float* b_dense = (const float*)d_in[2];
  const float* tau_n   = (const float*)d_in[3];
  const float* tau_m   = (const float*)d_in[4];
  const float* W_out   = (const float*)d_in[5];
  const float* b_out   = (const float*)d_in[6];
  float*       out     = (float*)d_out;

  char* ws = (char*)d_ws;
  _Float16* xh      = (_Float16*)(ws);
  _Float16* wh      = (_Float16*)(ws + (size_t)64  * 1024 * 1024);
  _Float16* dinh    = (_Float16*)(ws + (size_t)65  * 1024 * 1024);
  float*    partial = (float*)   (ws + (size_t)193 * 1024 * 1024);

  // 0) fp32 -> f16 conversions
  {
    int n4x = MTOT * II / 4;                       // 8,388,608
    cvt_f32_to_f16_x4<<<(n4x + 255) / 256, 256, 0, stream>>>(
        (const float4*)x, xh, n4x);
    int n4w = HH * II / 4;                          // 131,072
    cvt_f32_to_f16_x4<<<(n4w + 255) / 256, 256, 0, stream>>>(
        (const float4*)W_dense, wh, n4w);
  }

  // 1) WMMA GEMM: 2048 M-supertiles x 16 N-supertiles = 32768 waves
  {
    int nwaves = (MTOT / 32) * (HH / 64);
    int nblocks = nwaves / 8;                       // 8 waves / 256-thread block
    gemm_wmma_f16<<<nblocks, 256, 0, stream>>>(xh, wh, b_dense, dinh);
  }

  // 2) recurrence: one wave per (b, h-chunk)
  recurrence<<<BB * NCH, 32, 0, stream>>>(dinh, tau_n, tau_m, W_out, partial);

  // 3) reduce chunks + sigmoid
  finalize<<<MTOT / 256, 256, 0, stream>>>(partial, b_out, out);
}
